// OSRA_Attention_13391708029357
// MI455X (gfx1250) — matrix operand order repaired and checked by reading, not compiled, not run on hardware
//
#include <hip/hip_runtime.h>
#include <hip/hip_bf16.h>
#include <cstdint>

typedef __attribute__((ext_vector_type(16))) _Float16 v16h;
typedef __attribute__((ext_vector_type(8)))  float    v8f;

#define EPSV 1e-5f

// ---------------------------------------------------------------------------
// WMMA operand loaders (CDNA5 16x16x32 f16 layouts, wave32)
// ---------------------------------------------------------------------------

// A matrix 16x32 f16, row-major source with leading dim `ld` (elements).
// Per ISA: lanes 0-15 M=lane; V0-3 K=0..7 (half0) / 8..15 (half1); V4-7 K=16..23 / 24..31.
static __device__ __forceinline__ v16h load_a_rm(const _Float16* base, int ld) {
  const int lane = threadIdx.x & 31;
  const int half = lane >> 4;
  const int m    = lane & 15;
  v16h a;
#pragma unroll
  for (int j = 0; j < 8; ++j) {
    const int kb = ((j & 4) ? 16 : 0) + half * 8 + 2 * (j & 3);
    const _Float16* p = base + m * ld + kb;
    a[2 * j]     = p[0];
    a[2 * j + 1] = p[1];
  }
  return a;
}

// B matrix 32x16 f16; element (k,n) at base[n*ldn + k*ldk].
// Per ISA: lane(0-15)=col n; V0-3 K=0..7 (half0) / 8..15 (half1); V4-7 K=16..23 / 24..31.
static __device__ __forceinline__ v16h load_b_gen(const _Float16* base, int ldk, int ldn) {
  const int lane = threadIdx.x & 31;
  const int half = lane >> 4;
  const int n    = lane & 15;
  v16h b;
#pragma unroll
  for (int j = 0; j < 8; ++j) {
    const int k = ((j & 4) ? 16 : 0) + half * 8 + 2 * (j & 3);
    b[2 * j]     = base[n * ldn + k * ldk];
    b[2 * j + 1] = base[n * ldn + (k + 1) * ldk];
  }
  return b;
}

// ---------------------------------------------------------------------------
// Elementwise / conv kernels
// ---------------------------------------------------------------------------

__global__ void k_cvt_f16(const float* __restrict__ src, _Float16* __restrict__ dst, int n) {
  int i = blockIdx.x * blockDim.x + threadIdx.x;
  if (i < n) dst[i] = (_Float16)src[i];
}

// dw5x5 s2 p2 -> BN1 -> exact GELU -> *pw -> BN2 ; out t[b][c][32*32]
__global__ void k_sr(const float* __restrict__ x, const float* __restrict__ dw,
                     const float* __restrict__ g1, const float* __restrict__ b1,
                     const float* __restrict__ m1, const float* __restrict__ v1,
                     const float* __restrict__ pw,
                     const float* __restrict__ g2, const float* __restrict__ b2,
                     const float* __restrict__ m2, const float* __restrict__ v2,
                     float* __restrict__ t) {
  int idx = blockIdx.x * blockDim.x + threadIdx.x;   // B*C*1024
  if (idx >= 4 * 256 * 1024) return;
  int pix = idx & 1023;
  int c   = (idx >> 10) & 255;
  int b   = idx >> 18;
  int y = pix >> 5, xx = pix & 31;
  float acc = 0.f;
#pragma unroll
  for (int dy = 0; dy < 5; ++dy) {
    int h = 2 * y + dy - 2;
    if ((unsigned)h >= 64u) continue;
#pragma unroll
    for (int dx = 0; dx < 5; ++dx) {
      int w = 2 * xx + dx - 2;
      if ((unsigned)w >= 64u) continue;
      acc += x[(((size_t)b * 4096) + h * 64 + w) * 256 + c] * dw[c * 25 + dy * 5 + dx];
    }
  }
  float inv1 = g1[c] * rsqrtf(v1[c] + EPSV);
  acc = acc * inv1 + (b1[c] - m1[c] * inv1);
  acc = 0.5f * acc * (1.f + erff(acc * 0.70710678118654752f));   // exact GELU
  acc *= pw[c];
  float inv2 = g2[c] * rsqrtf(v2[c] + EPSV);
  acc = acc * inv2 + (b2[c] - m2[c] * inv2);
  t[idx] = acc;
}

// dw3x3 p1 + bias + residual ; store token-major f16 kv[b][n][c]
__global__ void k_local(const float* __restrict__ t, const float* __restrict__ lw,
                        const float* __restrict__ lb, _Float16* __restrict__ kvh) {
  int idx = blockIdx.x * blockDim.x + threadIdx.x;   // B*C*1024
  if (idx >= 4 * 256 * 1024) return;
  int pix = idx & 1023;
  int c   = (idx >> 10) & 255;
  int b   = idx >> 18;
  int y = pix >> 5, xx = pix & 31;
  const float* tc = t + ((size_t)(b * 256 + c)) * 1024;
  float acc = lb[c] + tc[pix];
#pragma unroll
  for (int dy = 0; dy < 3; ++dy) {
    int yy = y + dy - 1;
    if ((unsigned)yy >= 32u) continue;
#pragma unroll
    for (int dx = 0; dx < 3; ++dx) {
      int x2 = xx + dx - 1;
      if ((unsigned)x2 >= 32u) continue;
      acc += tc[yy * 32 + x2] * lw[c * 9 + dy * 3 + dx];
    }
  }
  kvh[((size_t)(b * 1024) + pix) * 256 + c] = (_Float16)acc;
}

// ---------------------------------------------------------------------------
// GEMM kernels (WMMA f16 -> f32)
// ---------------------------------------------------------------------------

// q[b,tok,o] = sum_c x[b,tok,c]*Wq[o,c] + bq[o]; store q as [b,h,tok,d] f16
__global__ void __launch_bounds__(128) k_qproj(const _Float16* __restrict__ xh,
                                               const _Float16* __restrict__ wqh,
                                               const float* __restrict__ bq,
                                               _Float16* __restrict__ qh) {
  const int wave = threadIdx.x >> 5;
  const int tile = blockIdx.x * 4 + wave;   // 1024 m-tiles * 16 n-tiles
  const int m0 = (tile >> 4) * 16;          // row over B*N
  const int o0 = (tile & 15) * 16;
  v8f acc = {};
#pragma unroll
  for (int kk = 0; kk < 256; kk += 32) {
    v16h a = load_a_rm(xh + (size_t)m0 * 256 + kk, 256);
    v16h b = load_b_gen(wqh + (size_t)o0 * 256 + kk, 1, 256);
    acc = __builtin_amdgcn_wmma_f32_16x16x32_f16(false, a, false, b, (short)0, acc, false, false);
  }
  const int lane = threadIdx.x & 31;
  const int half = lane >> 4;
  const int o = o0 + (lane & 15);
  const float bias = bq[o];
  const int h = o >> 5, d = o & 31;
#pragma unroll
  for (int j = 0; j < 8; ++j) {
    int row = m0 + j + half * 8;
    int b_  = row >> 12;
    int tok = row & 4095;
    qh[(((size_t)(b_ * 8 + h)) * 4096 + tok) * 32 + d] = (_Float16)(acc[j] + bias);
  }
}

// kvp[b,o,n] = sum_c Wkv[o,c]*kv[b,c,n] + bkv[o]
// k stored [b,h,n,d] f16 (key-major); v stored TRANSPOSED [b,h,d,n] f16 so the
// attention P.V B-operand reads two runs of 8 contiguous halves per lane (-> b128 loads).
__global__ void __launch_bounds__(128) k_kvproj(const _Float16* __restrict__ wkvh,
                                                const _Float16* __restrict__ kvh,
                                                const float* __restrict__ bkv,
                                                _Float16* __restrict__ kh,
                                                _Float16* __restrict__ vh) {
  const int wave = threadIdx.x >> 5;
  const int tile = blockIdx.x * 4 + wave;   // B * 32 * 64
  const int b_  = tile >> 11;
  const int rem = tile & 2047;
  const int o0 = (rem >> 6) * 16;
  const int n0 = (rem & 63) * 16;
  v8f acc = {};
#pragma unroll
  for (int kk = 0; kk < 256; kk += 32) {
    v16h a = load_a_rm(wkvh + (size_t)o0 * 256 + kk, 256);
    v16h b = load_b_gen(kvh + ((size_t)(b_ * 1024 + n0)) * 256 + kk, 1, 256);
    acc = __builtin_amdgcn_wmma_f32_16x16x32_f16(false, a, false, b, (short)0, acc, false, false);
  }
  const int lane = threadIdx.x & 31;
  const int half = lane >> 4;
  const int n = n0 + (lane & 15);
#pragma unroll
  for (int j = 0; j < 8; ++j) {
    int o = o0 + j + half * 8;
    float val = acc[j] + bkv[o];
    if (o < 256) {
      int h = o >> 5, d = o & 31;
      kh[(((size_t)(b_ * 8 + h)) * 1024 + n) * 32 + d] = (_Float16)val;           // [Nk, hd]
    } else {
      int oo = o - 256, h = oo >> 5, d = oo & 31;
      vh[((((size_t)(b_ * 8 + h)) * 32 + d)) * 1024 + n] = (_Float16)val;         // [hd, Nk]
    }
  }
}

// ---------------------------------------------------------------------------
// Fused attention: S = q·k^T*scale + rel; online softmax; O = P·v
// block = 4 waves, each wave owns 16 q-rows; loop keys in chunks of 32.
// ---------------------------------------------------------------------------
__global__ void __launch_bounds__(128) k_attn(const _Float16* __restrict__ qh,
                                              const _Float16* __restrict__ kh,
                                              const _Float16* __restrict__ vh,
                                              const float* __restrict__ rel,
                                              float* __restrict__ out) {
  __shared__ _Float16 Pbuf[4][16 * 32];
  const int wave = threadIdx.x >> 5;
  const int lane = threadIdx.x & 31;
  const int half = lane >> 4;
  const int col  = lane & 15;
  const int bh   = blockIdx.x >> 6;   // b*8+h
  const int mblk = blockIdx.x & 63;
  const int b_ = bh >> 3, h = bh & 7;
  const int row0 = mblk * 64 + wave * 16;

  const _Float16* qbase = qh + ((size_t)bh * 4096 + row0) * 32;
  const _Float16* kbase = kh + (size_t)bh * 1024 * 32;   // [Nk, hd]
  const _Float16* vbase = vh + (size_t)bh * 1024 * 32;   // [hd, Nk]
  const float* relbase  = rel + ((size_t)h * 4096 + row0) * 1024;

  const v16h aq = load_a_rm(qbase, 32);
  v8f acc0 = {}, acc1 = {};
  float mi[8], li[8];
#pragma unroll
  for (int j = 0; j < 8; ++j) { mi[j] = -1e30f; li[j] = 0.f; }
  const float scale = 0.17677669529663687f;   // 32^-0.5
  const v8f vzero = {};
  _Float16* pb = &Pbuf[wave][0];

  for (int kk = 0; kk < 1024; kk += 32) {
    // S chunk [16 x 32]: two WMMAs (k stored key-major so K-pairs contiguous)
    v16h bk0 = load_b_gen(kbase + (size_t)kk * 32, 1, 32);
    v16h bk1 = load_b_gen(kbase + (size_t)(kk + 16) * 32, 1, 32);
    v8f s0 = __builtin_amdgcn_wmma_f32_16x16x32_f16(false, aq, false, bk0, (short)0, vzero, false, false);
    v8f s1 = __builtin_amdgcn_wmma_f32_16x16x32_f16(false, aq, false, bk1, (short)0, vzero, false, false);
#pragma unroll
    for (int j = 0; j < 8; ++j) {
      const int m = j + half * 8;                       // row in tile
      const float* rp = relbase + (size_t)m * 1024 + kk + col;
      // prefetch the rel_pos stream two chunks ahead (global_prefetch_b8)
      __builtin_prefetch((const void*)(rp + 64), 0, 0);
      float e0 = s0[j] * scale + rp[0];
      float e1 = s1[j] * scale + rp[16];
      // row max across the 16-lane half (masks 1,2,4,8 stay in-half)
      float cm = fmaxf(e0, e1);
#pragma unroll
      for (int off = 1; off < 16; off <<= 1) cm = fmaxf(cm, __shfl_xor(cm, off, 32));
      float newm  = fmaxf(mi[j], cm);
      float alpha = __expf(mi[j] - newm);
      float p0 = __expf(e0 - newm);
      float p1 = __expf(e1 - newm);
      float rs = p0 + p1;
#pragma unroll
      for (int off = 1; off < 16; off <<= 1) rs += __shfl_xor(rs, off, 32);
      li[j] = li[j] * alpha + rs;
      mi[j] = newm;
      acc0[j] *= alpha;
      acc1[j] *= alpha;
      // stage P (C-layout -> row-major LDS, wave-private => no barrier needed)
      pb[m * 32 + col]      = (_Float16)p0;
      pb[m * 32 + 16 + col] = (_Float16)p1;
    }
    // P·V: A from LDS row-major; B from v^T, K-contiguous => b128 loads
    v16h ap  = load_a_rm(pb, 32);
    v16h bv0 = load_b_gen(vbase + kk, 1, 1024);                       // d = 0..15
    v16h bv1 = load_b_gen(vbase + (size_t)16 * 1024 + kk, 1, 1024);   // d = 16..31
    acc0 = __builtin_amdgcn_wmma_f32_16x16x32_f16(false, ap, false, bv0, (short)0, acc0, false, false);
    acc1 = __builtin_amdgcn_wmma_f32_16x16x32_f16(false, ap, false, bv1, (short)0, acc1, false, false);
  }
#pragma unroll
  for (int j = 0; j < 8; ++j) {
    int rg = row0 + j + half * 8;
    float inv = 1.f / li[j];
    size_t base = ((size_t)b_ * 4096 + rg) * 256 + h * 32;
    out[base + col]      = acc0[j] * inv;
    out[base + 16 + col] = acc1[j] * inv;
  }
}

// ---------------------------------------------------------------------------

extern "C" void kernel_launch(void* const* d_in, const int* in_sizes, int n_in,
                              void* d_out, int out_size, void* d_ws, size_t ws_size,
                              hipStream_t stream) {
  (void)in_sizes; (void)n_in; (void)out_size; (void)ws_size;
  const float* x   = (const float*)d_in[0];
  const float* rel = (const float*)d_in[1];
  const float* Wq  = (const float*)d_in[4];
  const float* bq  = (const float*)d_in[5];
  const float* Wkv = (const float*)d_in[6];
  const float* bkv = (const float*)d_in[7];
  const float* dw  = (const float*)d_in[8];
  const float* g1  = (const float*)d_in[9];
  const float* b1  = (const float*)d_in[10];
  const float* m1  = (const float*)d_in[11];
  const float* v1  = (const float*)d_in[12];
  const float* pw  = (const float*)d_in[13];
  const float* g2  = (const float*)d_in[14];
  const float* b2  = (const float*)d_in[15];
  const float* m2  = (const float*)d_in[16];
  const float* v2  = (const float*)d_in[17];
  const float* lw  = (const float*)d_in[18];
  const float* lb  = (const float*)d_in[19];
  float* out = (float*)d_out;

  char* ws = (char*)d_ws;
  float*    t_buf = (float*)(ws);                                 // 4 MB  [B,C,1024]
  _Float16* kvh   = (_Float16*)(ws + (4u << 20));                 // 2 MB  [B,Nk,C]
  _Float16* xh    = (_Float16*)(ws + (6u << 20));                 // 8 MB  [B*N,C]
  _Float16* wqh   = (_Float16*)(ws + (14u << 20));                // 128 KB
  _Float16* wkvh  = (_Float16*)(ws + (14u << 20) + (128u << 10)); // 256 KB
  _Float16* qh    = (_Float16*)(ws + (14u << 20) + (384u << 10)); // 8 MB  [B,nh,N,hd]
  _Float16* kh    = (_Float16*)(ws + (22u << 20) + (384u << 10)); // 2 MB  [B,nh,Nk,hd]
  _Float16* vh    = (_Float16*)(ws + (24u << 20) + (384u << 10)); // 2 MB  [B,nh,hd,Nk]

  k_cvt_f16<<<(4194304 + 255) / 256, 256, 0, stream>>>(x, xh, 4194304);
  k_cvt_f16<<<(65536 + 255) / 256, 256, 0, stream>>>(Wq, wqh, 65536);
  k_cvt_f16<<<(131072 + 255) / 256, 256, 0, stream>>>(Wkv, wkvh, 131072);

  k_sr<<<4096, 256, 0, stream>>>(x, dw, g1, b1, m1, v1, pw, g2, b2, m2, v2, t_buf);
  k_local<<<4096, 256, 0, stream>>>(t_buf, lw, lb, kvh);

  k_qproj<<<4096, 128, 0, stream>>>(xh, wqh, bq, qh);     // 16384 tiles / 4 waves
  k_kvproj<<<2048, 128, 0, stream>>>(wkvh, kvh, bkv, kh, vh);
  k_attn<<<2048, 128, 0, stream>>>(qh, kh, vh, rel, out); // (B*nh) * (N/64)
}
